// RDML_69415261438730
// MI455X (gfx1250) — matrix operand order repaired and checked by reading, not compiled, not run on hardware
//
#include <hip/hip_runtime.h>

// ---------------------------------------------------------------------------
// Problem constants (match reference)
// ---------------------------------------------------------------------------
#define B_  32768
#define D_  512
#define C_  10
#define V_  3
#define BC_ (B_ * C_)          // 327680 per output array

typedef __attribute__((ext_vector_type(16))) __bf16 v16bf;
typedef __attribute__((ext_vector_type(8)))  __bf16 v8bf;
typedef __attribute__((ext_vector_type(8)))  float  v8f;
typedef unsigned int v4u __attribute__((ext_vector_type(4)));
typedef int          v4i __attribute__((ext_vector_type(4)));
typedef int          v8i __attribute__((ext_vector_type(8)));

// D = A(16x32 bf16) * B(32x16 bf16) + C(16x16 f32)
#define WMMA_BF16(a, b, c) \
    __builtin_amdgcn_wmma_f32_16x16x32_bf16(false, (a), false, (b), (short)0, (c), false, false)

__device__ __forceinline__ float softplusf(float x) {
    return fmaxf(x, 0.0f) + log1pf(__expf(-fabsf(x)));
}
__device__ __forceinline__ float sigmoidf(float x) {
    return 1.0f / (1.0f + __expf(-x));
}

// A fragment (16x32 bf16) for row `rowp` of a row-major bf16 matrix, K-block k0.
// Per ISA layout: lane<16 holds K = k0+e (e<8) and k0+16+e; lane>=16 shifts by 8.
__device__ __forceinline__ v16bf load_a_bf16(const __bf16* rowp, int k0, int lh) {
    v8bf a0 = *(const v8bf*)(rowp + k0 + lh * 8);
    v8bf a1 = *(const v8bf*)(rowp + k0 + lh * 8 + 16);
    return __builtin_shufflevector(a0, a1, 0,1,2,3,4,5,6,7,8,9,10,11,12,13,14,15);
}

__device__ __forceinline__ v16bf cvt_a(v8f q0, v8f q1) {
    v16bf a;
#pragma unroll
    for (int e = 0; e < 8; ++e) { a[e] = (__bf16)q0[e]; a[8 + e] = (__bf16)q1[e]; }
    return a;
}

// A fragment from a row-major f32 matrix (convert to bf16 in-register).
__device__ __forceinline__ v16bf load_a_f32(const float* rowp, int k0, int lh) {
    v8f q0 = *(const v8f*)(rowp + k0 + lh * 8);
    v8f q1 = *(const v8f*)(rowp + k0 + lh * 8 + 16);
    return cvt_a(q0, q1);
}

// B fragment (32x16 bf16): B[k][n] = W[n][k], W row-major (rows contiguous in k).
// lane n = lane&15, K = k0 + 8*lh + e (e<8) and k0 + 16 + 8*lh + e.
__device__ __forceinline__ v16bf load_b(const __bf16* Wb, int n, int k0, int lh) {
    return load_a_bf16(Wb + n * D_, k0, lh);
}

// ---------------------------------------------------------------------------
// Weight conversion: W_rec -> bf16; W_ecn/W_aux/W_ev -> bf16 padded to 16 rows
// ---------------------------------------------------------------------------
__global__ void convert_weights(const float* __restrict__ Wrec,
                                const float* __restrict__ Wecn,
                                const float* __restrict__ Waux,
                                const float* __restrict__ Wev,
                                __bf16* __restrict__ WrecB,
                                __bf16* __restrict__ WecnP,
                                __bf16* __restrict__ WauxP,
                                __bf16* __restrict__ WevP) {
    const int NREC = V_ * D_ * D_;            // 786432
    const int NPAD = V_ * 16 * D_;            // 24576 per padded array
    int idx = blockIdx.x * 256 + threadIdx.x;
    if (idx < NREC) { WrecB[idx] = (__bf16)Wrec[idx]; return; }
    int j = idx - NREC;
    if (j >= 3 * NPAD) return;
    int arr = j / NPAD; j %= NPAD;
    int v = j / (16 * D_);
    int rem = j % (16 * D_);
    int r = rem / D_, k = rem % D_;
    const float* src = (arr == 0) ? Wecn : (arr == 1) ? Waux : Wev;
    __bf16* dst      = (arr == 0) ? WecnP : (arr == 1) ? WauxP : WevP;
    float val = (r < C_) ? src[(v * C_ + r) * D_ + k] : 0.0f;
    dst[(v * 16 + r) * D_ + k] = (__bf16)val;
}

// ---------------------------------------------------------------------------
// Generic (Bx512)x(512x16) WMMA GEMM, f32 A, epilogue softplus -> ev buffer
// One wave per 16-row M tile. grid = 256 blocks x 256 threads (8 waves).
// ---------------------------------------------------------------------------
__global__ void gemm_sp_cd(const float* __restrict__ A, const __bf16* __restrict__ Wb,
                           const float* __restrict__ bias, float* __restrict__ evb) {
    const int lane = threadIdx.x & 31;
    const int mt = blockIdx.x * 8 + (threadIdx.x >> 5);
    const int lh = lane >> 4, mr = lane & 15;
    const int rowBase = mt * 16;
    const float* rowp = A + (size_t)(rowBase + mr) * D_;
    v8f acc = {0, 0, 0, 0, 0, 0, 0, 0};
    for (int k0 = 0; k0 < D_; k0 += 32) {
        v16bf a = load_a_f32(rowp, k0, lh);
        v16bf b = load_b(Wb, mr, k0, lh);
        acc = WMMA_BF16(a, b, acc);
    }
    float bv = (mr < C_) ? bias[mr] : 0.0f;
#pragma unroll
    for (int r = 0; r < 8; ++r) {
        int row = rowBase + lh * 8 + r;
        evb[row * 16 + mr] = (mr < C_) ? softplusf(acc[r] + bv) : 0.0f;
    }
}

// ---------------------------------------------------------------------------
// front_b: rob = ev @ W_sep^T + b_sep -> mask -> om, nr_bf16 = f*om
// ---------------------------------------------------------------------------
__global__ void front_b(const float* __restrict__ evb, const float* __restrict__ Wsep,
                        const float* __restrict__ bsep, const float* __restrict__ f,
                        float* __restrict__ om, __bf16* __restrict__ nr) {
    int idx = blockIdx.x * 256 + threadIdx.x;
    int row = idx >> 9, d = idx & (D_ - 1);
    float rob = bsep[d];
#pragma unroll
    for (int c = 0; c < C_; ++c)
        rob += evb[row * 16 + c] * Wsep[d * C_ + c];
    float m = sigmoidf(rob);
    const float P = 1e-8f;
    const float INV = 1.0f / (0.1f + 1e-8f);   // 1/(TAU+P); Gumbel noise cancels in a-r
    float ar = (logf(m + P) - logf(1.0f - m + P)) * INV;
    float mask = sigmoidf(ar);
    float omv = 1.0f - mask;
    om[idx] = omv;
    nr[idx] = (__bf16)(f[idx] * omv);
}

// ---------------------------------------------------------------------------
// rec GEMM: (Bx512) @ W_rec^T (512x512), bf16 WMMA.
// Each block: 8 waves share one 64-col W slice; the 64KB slice is staged into
// LDS once via the Tensor Data Mover (TDM), then B fragments come from LDS.
// Fused epilogue: feat = f*(1-om) + softplus(acc + b_rec)*om
// grid = 2048 blocks (256 M-blocks x 8 N-groups) x 256 threads.
// ---------------------------------------------------------------------------
__global__ void rec_gemm(const __bf16* __restrict__ nr, const __bf16* __restrict__ WrecB,
                         const float* __restrict__ brec, const float* __restrict__ f,
                         const float* __restrict__ om, float* __restrict__ feat) {
    __shared__ __bf16 Wlds[64 * D_];          // 64KB: rows [colBase, colBase+64) of W
    const int tid = threadIdx.x;
    const int lane = tid & 31;
    const int wav = tid >> 5;
    const int mtb = blockIdx.x & 255;
    const int ng  = blockIdx.x >> 8;
    const int colBase = ng * 64;

    // --- TDM stage: wave 0 issues one tensor_load_to_lds for the whole slice ---
    if (wav == 0) {
        unsigned long long gaddr =
            (unsigned long long)(uintptr_t)(WrecB + (size_t)colBase * D_);
        unsigned ldsoff = (unsigned)(uintptr_t)(&Wlds[0]);
        v4u g0;
        g0[0] = 1u;                                        // count=1 (valid user D#)
        g0[1] = ldsoff;                                    // lds_addr (bytes)
        g0[2] = (unsigned)(gaddr & 0xFFFFFFFFu);           // global_addr[31:0]
        g0[3] = (unsigned)((gaddr >> 32) & 0x1FFFFFFu)     // global_addr[56:32]
                | (2u << 30);                              // type = 2 ("image")
        v8i g1;
        g1[0] = (int)(1u << 16);                           // data_size = 1 -> 2 bytes
        g1[1] = (int)(((unsigned)D_ & 0xFFFFu) << 16);     // tensor_dim0 = 512 (lo16)
        g1[2] = (int)(((unsigned)D_ >> 16) | (64u << 16)); // dim0 hi | tensor_dim1 = 64
        g1[3] = (int)(((unsigned)D_) << 16);               // dim1 hi(0) | tile_dim0 = 512
        g1[4] = 64;                                        // tile_dim1 = 64, tile_dim2 = 0
        g1[5] = D_;                                        // tensor_dim0_stride = 512
        g1[6] = 0;
        g1[7] = 0;
        v4i gz4 = {0, 0, 0, 0};
        v8i gz8 = {0, 0, 0, 0, 0, 0, 0, 0};
        __builtin_amdgcn_tensor_load_to_lds(g0, g1, gz4, gz4, gz8, 0);
        __builtin_amdgcn_s_wait_tensorcnt(0);
    }
    __syncthreads();

    const int lh = lane >> 4, mr = lane & 15;
    const int rowBase = (mtb * 8 + wav) * 16;
    const __bf16* rowp = nr + (size_t)(rowBase + mr) * D_;
    const __bf16* lb = &Wlds[0];
    v8f acc0 = {0,0,0,0,0,0,0,0}, acc1 = acc0, acc2 = acc0, acc3 = acc0;
    for (int k0 = 0; k0 < D_; k0 += 32) {
        v16bf a = load_a_bf16(rowp, k0, lh);
        v16bf b0 = load_a_bf16(lb + (0 * 16 + mr) * D_, k0, lh);
        v16bf b1 = load_a_bf16(lb + (1 * 16 + mr) * D_, k0, lh);
        v16bf b2 = load_a_bf16(lb + (2 * 16 + mr) * D_, k0, lh);
        v16bf b3 = load_a_bf16(lb + (3 * 16 + mr) * D_, k0, lh);
        acc0 = WMMA_BF16(a, b0, acc0);
        acc1 = WMMA_BF16(a, b1, acc1);
        acc2 = WMMA_BF16(a, b2, acc2);
        acc3 = WMMA_BF16(a, b3, acc3);
    }
#pragma unroll
    for (int j = 0; j < 4; ++j) {
        v8f acc = (j == 0) ? acc0 : (j == 1) ? acc1 : (j == 2) ? acc2 : acc3;
        int col = colBase + j * 16 + mr;
        float bv = brec[col];
#pragma unroll
        for (int r = 0; r < 8; ++r) {
            int row = rowBase + lh * 8 + r;
            size_t off = (size_t)row * D_ + col;
            float omv = om[off];
            float fv = f[off];
            feat[off] = fv * (1.0f - omv) + softplusf(acc[r] + bv) * omv;
        }
    }
}

// ---------------------------------------------------------------------------
// aux: r_out/nr_out/rec_out = sp( {r,nr,rec}_feat @ W_aux^T + b_aux )
// fragments built on the fly: r=f*(1-om), nr=f*om, rec=feat-r
// ---------------------------------------------------------------------------
__global__ void aux_gemm(const float* __restrict__ f, const float* __restrict__ om,
                         const float* __restrict__ feat, const __bf16* __restrict__ Wb,
                         const float* __restrict__ bias, float* __restrict__ out_r,
                         float* __restrict__ out_nr, float* __restrict__ out_rc) {
    const int lane = threadIdx.x & 31;
    const int mt = blockIdx.x * 8 + (threadIdx.x >> 5);
    const int lh = lane >> 4, mr = lane & 15;
    const int rowBase = mt * 16;
    const size_t abase = (size_t)(rowBase + mr) * D_;
    const float* fp = f + abase;
    const float* op = om + abase;
    const float* tp = feat + abase;
    v8f accR = {0,0,0,0,0,0,0,0}, accN = accR, accC = accR;
    for (int k0 = 0; k0 < D_; k0 += 32) {
        int o0 = k0 + lh * 8, o1 = o0 + 16;
        v8f f0 = *(const v8f*)(fp + o0), f1 = *(const v8f*)(fp + o1);
        v8f m0 = *(const v8f*)(op + o0), m1 = *(const v8f*)(op + o1);
        v8f t0 = *(const v8f*)(tp + o0), t1 = *(const v8f*)(tp + o1);
        v8f r0 = f0 - f0 * m0, r1 = f1 - f1 * m1;       // f*(1-om)
        v16bf aR = cvt_a(r0, r1);
        v16bf aN = cvt_a(f0 * m0, f1 * m1);             // f*om
        v16bf aC = cvt_a(t0 - r0, t1 - r1);             // feat - r_feat
        v16bf b = load_b(Wb, mr, k0, lh);
        accR = WMMA_BF16(aR, b, accR);
        accN = WMMA_BF16(aN, b, accN);
        accC = WMMA_BF16(aC, b, accC);
    }
    if (mr < C_) {
        float bv = bias[mr];
#pragma unroll
        for (int r = 0; r < 8; ++r) {
            int row = rowBase + lh * 8 + r;
            out_r [row * C_ + mr] = softplusf(accR[r] + bv);
            out_nr[row * C_ + mr] = softplusf(accN[r] + bv);
            out_rc[row * C_ + mr] = softplusf(accC[r] + bv);
        }
    }
}

// ---------------------------------------------------------------------------
// fus_b: att = softmax(evb row), g = sp(att @ W_ce^T + b_ce) * feat  (bf16 out)
// ---------------------------------------------------------------------------
__global__ void fus_b(const float* __restrict__ evb, const float* __restrict__ Wce,
                      const float* __restrict__ bce, const float* __restrict__ feat,
                      __bf16* __restrict__ g) {
    int idx = blockIdx.x * 256 + threadIdx.x;
    int row = idx >> 9, d = idx & (D_ - 1);
    float e[C_];
    float mx = -1e30f;
#pragma unroll
    for (int c = 0; c < C_; ++c) { e[c] = evb[row * 16 + c]; mx = fmaxf(mx, e[c]); }
    float s = 0.0f;
#pragma unroll
    for (int c = 0; c < C_; ++c) { e[c] = __expf(e[c] - mx); s += e[c]; }
    float inv = 1.0f / s;
    float gp = bce[d];
#pragma unroll
    for (int c = 0; c < C_; ++c) gp += (e[c] * inv) * Wce[d * C_ + c];
    g[idx] = (__bf16)(softplusf(gp) * feat[idx]);
}

// ---------------------------------------------------------------------------
// fus_c: out = sp(g @ W_ev^T + b_ev)  -- bf16 A, WMMA
// ---------------------------------------------------------------------------
__global__ void fus_c(const __bf16* __restrict__ g, const __bf16* __restrict__ Wb,
                      const float* __restrict__ bias, float* __restrict__ outv) {
    const int lane = threadIdx.x & 31;
    const int mt = blockIdx.x * 8 + (threadIdx.x >> 5);
    const int lh = lane >> 4, mr = lane & 15;
    const int rowBase = mt * 16;
    const __bf16* rowp = g + (size_t)(rowBase + mr) * D_;
    v8f acc = {0,0,0,0,0,0,0,0};
    for (int k0 = 0; k0 < D_; k0 += 32) {
        v16bf a = load_a_bf16(rowp, k0, lh);
        v16bf b = load_b(Wb, mr, k0, lh);
        acc = WMMA_BF16(a, b, acc);
    }
    if (mr < C_) {
        float bv = bias[mr];
#pragma unroll
        for (int r = 0; r < 8; ++r) {
            int row = rowBase + lh * 8 + r;
            outv[row * C_ + mr] = softplusf(acc[r] + bv);
        }
    }
}

// ---------------------------------------------------------------------------
// Dempster-Shafer combination over the 3 view outputs (one thread per row)
// ---------------------------------------------------------------------------
__global__ void ds_combine(const float* __restrict__ outs, float* __restrict__ evd) {
    int row = blockIdx.x * 256 + threadIdx.x;
    if (row >= B_) return;
    float alpha[C_];
#pragma unroll
    for (int c = 0; c < C_; ++c) alpha[c] = outs[row * C_ + c] + 1.0f;
    for (int v = 1; v < V_; ++v) {
        const float* o = outs + (size_t)v * BC_ + row * C_;
        float a2[C_];
#pragma unroll
        for (int c = 0; c < C_; ++c) a2[c] = o[c] + 1.0f;
        float S1 = 0.0f, S2 = 0.0f;
#pragma unroll
        for (int c = 0; c < C_; ++c) { S1 += alpha[c]; S2 += a2[c]; }
        float u1 = (float)C_ / S1, u2 = (float)C_ / S2;
        float b1[C_], b2[C_];
        float sb1 = 0.0f, sb2 = 0.0f, dot = 0.0f;
#pragma unroll
        for (int c = 0; c < C_; ++c) {
            b1[c] = (alpha[c] - 1.0f) / S1;
            b2[c] = (a2[c] - 1.0f) / S2;
            sb1 += b1[c]; sb2 += b2[c]; dot += b1[c] * b2[c];
        }
        float kk = 1.0f - (sb1 * sb2 - dot);
        float u_a = u1 * u2 / kk;
        float S_a = (float)C_ / u_a;
#pragma unroll
        for (int c = 0; c < C_; ++c)
            alpha[c] = ((b1[c] * b2[c] + b1[c] * u2 + b2[c] * u1) / kk) * S_a + 1.0f;
    }
#pragma unroll
    for (int c = 0; c < C_; ++c) evd[row * C_ + c] = alpha[c] - 1.0f;
}

// ---------------------------------------------------------------------------
// Host launch
// ---------------------------------------------------------------------------
extern "C" void kernel_launch(void* const* d_in, const int* in_sizes, int n_in,
                              void* d_out, int out_size, void* d_ws, size_t ws_size,
                              hipStream_t stream) {
    (void)in_sizes; (void)n_in; (void)out_size; (void)ws_size;
    const float* x[3] = { (const float*)d_in[0], (const float*)d_in[1], (const float*)d_in[2] };
    const float* W_sep = (const float*)d_in[3];
    const float* b_sep = (const float*)d_in[4];
    const float* W_rec = (const float*)d_in[5];
    const float* b_rec = (const float*)d_in[6];
    const float* W_aux = (const float*)d_in[7];
    const float* b_aux = (const float*)d_in[8];
    const float* W_ce  = (const float*)d_in[9];
    const float* b_ce  = (const float*)d_in[10];
    const float* W_ev  = (const float*)d_in[11];
    const float* b_ev  = (const float*)d_in[12];
    const float* W_ecn = (const float*)d_in[13];
    const float* b_ecn = (const float*)d_in[14];
    float* out = (float*)d_out;

    // Workspace layout (bytes, all 256-aligned)
    char* ws = (char*)d_ws;
    __bf16* WrecB = (__bf16*)(ws + 0);                         // 3*512*512*2 = 1572864
    __bf16* WecnP = (__bf16*)(ws + 1572864);                   // 3*16*512*2  = 49152
    __bf16* WauxP = (__bf16*)(ws + 1622016);                   // 49152
    __bf16* WevP  = (__bf16*)(ws + 1671168);                   // 49152
    float*  evb   = (float*)(ws + 1720320);                    // B*16*4      = 2097152
    float*  omb   = (float*)(ws + 3817472);                    // B*512*4     = 67108864
    float*  featb = (float*)(ws + 70926336);                   // 67108864
    __bf16* gb    = (__bf16*)(ws + 138035200);                 // B*512*2     = 33554432
    // total: 171589632 bytes

    convert_weights<<<3360, 256, 0, stream>>>(W_rec, W_ecn, W_aux, W_ev,
                                              WrecB, WecnP, WauxP, WevP);

    const int GEMM_BLK = (B_ / 16) / 8;   // 256 blocks, one wave per 16-row tile
    const int EW_BLK   = (B_ * D_) / 256; // 65536 blocks for elementwise passes

    for (int v = 0; v < V_; ++v) {
        const float* f = x[v];
        const __bf16* WrecB_v = WrecB + (size_t)v * D_ * D_;
        const __bf16* WecnP_v = WecnP + (size_t)v * 16 * D_;
        const __bf16* WauxP_v = WauxP + (size_t)v * 16 * D_;
        const __bf16* WevP_v  = WevP  + (size_t)v * 16 * D_;
        const float* Wsep_v = W_sep + (size_t)v * D_ * C_;
        const float* bsep_v = b_sep + (size_t)v * D_;
        const float* brec_v = b_rec + (size_t)v * D_;
        const float* baux_v = b_aux + (size_t)v * C_;
        const float* Wce_v  = W_ce  + (size_t)v * D_ * C_;
        const float* bce_v  = b_ce  + (size_t)v * D_;
        const float* bev_v  = b_ev  + (size_t)v * C_;
        const float* becn_v = b_ecn + (size_t)v * C_;

        float* out_v  = out + (size_t)(0 + v) * BC_;
        float* rout_v = out + (size_t)(3 + v) * BC_;
        float* nout_v = out + (size_t)(6 + v) * BC_;
        float* cout_v = out + (size_t)(9 + v) * BC_;

        // ev = sp(f @ W_ecn^T + b_ecn)
        gemm_sp_cd<<<GEMM_BLK, 256, 0, stream>>>(f, WecnP_v, becn_v, evb);
        // mask -> om, nr_bf16
        front_b<<<EW_BLK, 256, 0, stream>>>(evb, Wsep_v, bsep_v, f, omb, gb);
        // rec GEMM (TDM-staged W slice in LDS) + fused feat epilogue
        rec_gemm<<<2048, 256, 0, stream>>>(gb, WrecB_v, brec_v, f, omb, featb);
        // r_out / nr_out / rec_out
        aux_gemm<<<GEMM_BLK, 256, 0, stream>>>(f, omb, featb, WauxP_v, baux_v,
                                               rout_v, nout_v, cout_v);
        // att pre-softmax = sp(feat @ W_ecn^T + b_ecn)
        gemm_sp_cd<<<GEMM_BLK, 256, 0, stream>>>(featb, WecnP_v, becn_v, evb);
        // g = sp(att @ W_ce^T + b_ce) * feat  (bf16, reuse gb)
        fus_b<<<EW_BLK, 256, 0, stream>>>(evb, Wce_v, bce_v, featb, gb);
        // out = sp(g @ W_ev^T + b_ev)
        fus_c<<<GEMM_BLK, 256, 0, stream>>>(gb, WevP_v, bev_v, out_v);
    }

    // Dempster-Shafer combine -> evidence_a
    ds_combine<<<B_ / 256, 256, 0, stream>>>(out, out + (size_t)12 * BC_);
}
